// OmniAudioVQBridgeTokenizer_67044439490980
// MI455X (gfx1250) — compile-verified, matrix operand order checked by reading, not run on hardware
//
#include <hip/hip_runtime.h>
#include <hip/hip_bf16.h>
#include <math.h>

// ---------------------------------------------------------------------------
// OmniAudio VQ bridge tokenizer for MI455X (gfx1250, wave32, WMMA bf16,
// async global->LDS double-buffered GEMM staging)
// ---------------------------------------------------------------------------

typedef __attribute__((ext_vector_type(16))) __bf16 v16bf;
typedef __attribute__((ext_vector_type(8)))  __bf16 v8bf;
typedef __attribute__((ext_vector_type(8)))  float  v8f;

namespace {
constexpr int kM   = 4096;   // B * FRAMES
constexpr int kK   = 5120;   // I == P*D
constexpr int kN   = 5120;   // intermediate dim
constexpr int kD   = 1280;
constexpr int kP   = 4;
constexpr int kB   = 8;
constexpr int kF   = 512;
constexpr int kCBtot = 24576;
constexpr int kMaxChunks = 32; // fixed stride for chunk-min scratch
}

// ---------------------------------------------------------------------------
// CDNA5 async global->LDS copies (tracked by ASYNCcnt).
// VDST operand = per-lane LDS byte address (low 32 bits of flat LDS pointer),
// VADDR = 64-bit global address ("GV" mode, saddr=off).
// ---------------------------------------------------------------------------
__device__ __forceinline__ void async_copy_b128(void* lds, const void* g) {
#if defined(__HIP_DEVICE_COMPILE__)
    unsigned la = (unsigned)(unsigned long long)lds;      // addr[31:0] = LDS offset
    unsigned long long ga = (unsigned long long)g;
    asm volatile("global_load_async_to_lds_b128 %0, %1, off"
                 :: "v"(la), "v"(ga) : "memory");
#else
    (void)lds; (void)g;
#endif
}

__device__ __forceinline__ void async_copy_b64(void* lds, const void* g) {
#if defined(__HIP_DEVICE_COMPILE__)
    unsigned la = (unsigned)(unsigned long long)lds;
    unsigned long long ga = (unsigned long long)g;
    asm volatile("global_load_async_to_lds_b64 %0, %1, off"
                 :: "v"(la), "v"(ga) : "memory");
#else
    (void)lds; (void)g;
#endif
}

__device__ __forceinline__ void wait_async0() {
#if defined(__HIP_DEVICE_COMPILE__)
#if __has_builtin(__builtin_amdgcn_s_wait_asynccnt)
    __builtin_amdgcn_s_wait_asynccnt(0);
#else
    asm volatile("s_wait_asynccnt 0" ::: "memory");
#endif
#endif
}

// ---------------------------------------------------------------------------
// Elementwise helpers
// ---------------------------------------------------------------------------

__global__ void __launch_bounds__(256)
cvt_f32_bf16(const float* __restrict__ in, __bf16* __restrict__ out, long n) {
    long i = (long)blockIdx.x * blockDim.x + threadIdx.x;
    long stride = (long)gridDim.x * blockDim.x;
    for (; i < n; i += stride) out[i] = (__bf16)in[i];
}

// w[o, d, k]  ->  Bt[o, k*D + d]   (conv1d(kernel=stride=P) as GEMM weight)
__global__ void __launch_bounds__(256)
pack_conv_w(const float* __restrict__ w, __bf16* __restrict__ bt) {
    long n = (long)kN * kD * kP;
    long i = (long)blockIdx.x * blockDim.x + threadIdx.x;
    long stride = (long)gridDim.x * blockDim.x;
    for (; i < n; i += stride) {
        long o = i / ((long)kD * kP);
        long r = i % ((long)kD * kP);
        long d = r / kP;
        long k = r % kP;
        bt[o * (long)kK + k * kD + d] = (__bf16)w[i];
    }
}

__device__ __forceinline__ float block_sum256(float v, float* red) {
    int t = threadIdx.x;
    red[t] = v;
    __syncthreads();
    #pragma unroll
    for (int s = 128; s > 0; s >>= 1) {
        if (t < s) red[t] += red[t + s];
        __syncthreads();
    }
    float r = red[0];
    __syncthreads();
    return r;
}

__global__ void __launch_bounds__(256)
row_norm2(const float* __restrict__ cb, float* __restrict__ e2, int K) {
    __shared__ float red[256];
    long c = blockIdx.x;
    float s = 0.f;
    for (int i = threadIdx.x; i < K; i += 256) {
        float v = cb[c * (long)K + i];
        s += v * v;
    }
    s = block_sum256(s, red);
    if (threadIdx.x == 0) e2[c] = s;
}

__global__ void __launch_bounds__(256)
silu_mul(const float* __restrict__ g, const float* __restrict__ u,
         __bf16* __restrict__ h, long n) {
    long i = (long)blockIdx.x * blockDim.x + threadIdx.x;
    long stride = (long)gridDim.x * blockDim.x;
    for (; i < n; i += stride) {
        float gv = g[i];
        float hv = gv / (1.f + expf(-gv)) * u[i];
        h[i] = (__bf16)hv;
    }
}

// res = LN(c + x) ; write f32 residual and bf16 copy (WMMA A-operand)
__global__ void __launch_bounds__(256)
add_layernorm(const float* __restrict__ cmat, const float* __restrict__ x,
              const float* __restrict__ scale, const float* __restrict__ bias,
              float* __restrict__ resf, __bf16* __restrict__ resbf, int K) {
    __shared__ float red[256];
    long m = blockIdx.x;
    const float* cr = cmat + m * (long)K;
    const float* xr = x    + m * (long)K;
    float s = 0.f;
    for (int i = threadIdx.x; i < K; i += 256) s += cr[i] + xr[i];
    float mu = block_sum256(s, red) / (float)K;
    float v = 0.f;
    for (int i = threadIdx.x; i < K; i += 256) {
        float d = (cr[i] + xr[i]) - mu;
        v += d * d;
    }
    float var = block_sum256(v, red) / (float)K;
    float rstd = rsqrtf(var + 1e-5f);
    for (int i = threadIdx.x; i < K; i += 256) {
        float h = ((cr[i] + xr[i]) - mu) * rstd * scale[i] + bias[i];
        resf[m * (long)K + i] = h;
        resbf[m * (long)K + i] = (__bf16)h;
    }
}

// ---------------------------------------------------------------------------
// WMMA bf16 GEMM core: A[M,K] row-major bf16, Bt[N,K] row-major bf16.
// Block = 256 threads = 8 waves; block tile 32(M) x 256(N); K-step 32.
// Wave (wm in 0..1, wn in 0..3) owns a 16x64 tile = 4 WMMA accumulators.
// Double-buffered LDS tiles filled by async global->LDS DMA (ASYNCcnt),
// fragments per CDNA5 ISA 7.12.2 (16-bit A 16x32 / B 32x16).
// ---------------------------------------------------------------------------

#define LDS_STRIDE 40  // 32 bf16 + 8 pad (80B rows, spreads LDS banks)
#define A_TILE_ELEMS (32 * LDS_STRIDE)       // 1280 bf16 per buffer
#define B_TILE_ELEMS (256 * LDS_STRIDE)      // 10240 bf16 per buffer
#define SMEM_BYTES   ((2 * A_TILE_ELEMS + 2 * B_TILE_ELEMS) * 2)  // 46080 B

__device__ __forceinline__ void issue_tile_loads(
    const __bf16* __restrict__ A, const __bf16* __restrict__ Bt, int K,
    long m0, long n0, int k0, __bf16* As, __bf16* Bs, int t) {
    // A tile 32x32 bf16: 256 threads x 8 B (async b64)
    int row = t >> 3;
    int ch  = t & 7;
    async_copy_b64(As + row * LDS_STRIDE + ch * 4,
                   A + (m0 + row) * (long)K + k0 + ch * 4);
    // B tile 256x32 bf16: one row (= one N) per thread, 4x async b128
    const __bf16* src = Bt + (n0 + t) * (long)K + k0;
    #pragma unroll
    for (int j = 0; j < 4; ++j)
        async_copy_b128(Bs + t * LDS_STRIDE + j * 8, src + j * 8);
    if (k0 + 32 < K) __builtin_prefetch(src + 32, 0, 3); // global_prefetch_b8
}

__device__ __forceinline__ v16bf frag16(const __bf16* p) {
    const v8bf f0 = *(const v8bf*)(p);
    const v8bf f1 = *(const v8bf*)(p + 16);
    return __builtin_shufflevector(f0, f1, 0, 1, 2, 3, 4, 5, 6, 7,
                                           8, 9, 10, 11, 12, 13, 14, 15);
}

__device__ __forceinline__ void gemm_block_loop(
    const __bf16* __restrict__ A, const __bf16* __restrict__ Bt, int K,
    long m0, long n0, __bf16* smem_bf, v8f acc[4]) {
    const int t    = threadIdx.x;
    const int wave = t >> 5;
    const int lane = t & 31;
    const int wm   = wave >> 2;
    const int wn   = wave & 3;
    const int kb   = (lane < 16) ? 0 : 8;       // half-wave K-base
    const int rA   = wm * 16 + (lane & 15);     // A row within block tile

    __bf16* AsBuf[2] = {smem_bf, smem_bf + A_TILE_ELEMS};
    __bf16* BsBuf[2] = {smem_bf + 2 * A_TILE_ELEMS,
                        smem_bf + 2 * A_TILE_ELEMS + B_TILE_ELEMS};

    // prologue: start DMA of tile 0
    issue_tile_loads(A, Bt, K, m0, n0, 0, AsBuf[0], BsBuf[0], t);

    int buf = 0;
    for (int k0 = 0; k0 < K; k0 += 32, buf ^= 1) {
        wait_async0();       // this wave's DMA for `buf` is in LDS
        __syncthreads();     // every wave waited -> tile complete; prev reads done
        if (k0 + 32 < K)
            issue_tile_loads(A, Bt, K, m0, n0, k0 + 32,
                             AsBuf[buf ^ 1], BsBuf[buf ^ 1], t);

        const __bf16* Asb = AsBuf[buf];
        const __bf16* Bsb = BsBuf[buf];
        // preload all fragments, then issue 4 back-to-back WMMAs
        v16bf a = frag16(Asb + rA * LDS_STRIDE + kb);
        v16bf b0 = frag16(Bsb + (wn * 64 +  0 + (lane & 15)) * LDS_STRIDE + kb);
        v16bf b1 = frag16(Bsb + (wn * 64 + 16 + (lane & 15)) * LDS_STRIDE + kb);
        v16bf b2 = frag16(Bsb + (wn * 64 + 32 + (lane & 15)) * LDS_STRIDE + kb);
        v16bf b3 = frag16(Bsb + (wn * 64 + 48 + (lane & 15)) * LDS_STRIDE + kb);
#if defined(__HIP_DEVICE_COMPILE__)
        acc[0] = __builtin_amdgcn_wmma_f32_16x16x32_bf16(
            false, a, false, b0, (short)0, acc[0], false, false);
        acc[1] = __builtin_amdgcn_wmma_f32_16x16x32_bf16(
            false, a, false, b1, (short)0, acc[1], false, false);
        acc[2] = __builtin_amdgcn_wmma_f32_16x16x32_bf16(
            false, a, false, b2, (short)0, acc[2], false, false);
        acc[3] = __builtin_amdgcn_wmma_f32_16x16x32_bf16(
            false, a, false, b3, (short)0, acc[3], false, false);
#else
        (void)a; (void)b0; (void)b1; (void)b2; (void)b3;
#endif
    }
}

__global__ void __launch_bounds__(256)
gemm_bf16(const __bf16* __restrict__ A, const __bf16* __restrict__ Bt,
          float* __restrict__ C, int K, int N) {
    __shared__ __align__(16) char smem[SMEM_BYTES];
    long n0 = (long)blockIdx.x * 256;
    long m0 = (long)blockIdx.y * 32;

    const v8f zero = {0.f, 0.f, 0.f, 0.f, 0.f, 0.f, 0.f, 0.f};
    v8f acc[4] = {zero, zero, zero, zero};
    gemm_block_loop(A, Bt, K, m0, n0, (__bf16*)smem, acc);

    const int t = threadIdx.x, wave = t >> 5, lane = t & 31;
    const int wm = wave >> 2, wn = wave & 3;
    long rowBase = m0 + wm * 16 + ((lane < 16) ? 0 : 8);
    #pragma unroll
    for (int c = 0; c < 4; ++c) {
        long col = n0 + wn * 64 + c * 16 + (lane & 15);
        #pragma unroll
        for (int r = 0; r < 8; ++r)
            C[(rowBase + r) * (long)N + col] = acc[c][r];
    }
}

// ---------------------------------------------------------------------------
// VQ stage 1: same WMMA core, epilogue = per-row argmin over a 256-code chunk
// of key(m,c) = ||e_c||^2 - 2 * r_m . e_c  (||r||^2 is row-constant: dropped)
// key buffer aliases the (no longer needed) GEMM tile memory.
// ---------------------------------------------------------------------------
__global__ void __launch_bounds__(256)
vq_stage1(const __bf16* __restrict__ A, const __bf16* __restrict__ Bt,
          const float* __restrict__ e2, float* __restrict__ chunkVal,
          int* __restrict__ chunkIdx, int K) {
    __shared__ __align__(16) char smem[SMEM_BYTES];

    long n0 = (long)blockIdx.x * 256;
    long m0 = (long)blockIdx.y * 32;

    const v8f zero = {0.f, 0.f, 0.f, 0.f, 0.f, 0.f, 0.f, 0.f};
    v8f acc[4] = {zero, zero, zero, zero};
    gemm_block_loop(A, Bt, K, m0, n0, (__bf16*)smem, acc);

    float (*keybuf)[256] = (float (*)[256])smem;  // 32 KB, reuse tile LDS
    const int t = threadIdx.x, wave = t >> 5, lane = t & 31;
    const int wm = wave >> 2, wn = wave & 3;
    __syncthreads();  // all waves done reading tiles before we overwrite
    #pragma unroll
    for (int c = 0; c < 4; ++c) {
        int col = wn * 64 + c * 16 + (lane & 15);
        float en = e2[n0 + col];
        #pragma unroll
        for (int r = 0; r < 8; ++r) {
            int row = wm * 16 + r + ((lane < 16) ? 0 : 8);
            keybuf[row][col] = en - 2.f * acc[c][r];
        }
    }
    __syncthreads();
    if (t < 32) {
        float best = keybuf[t][0];
        int bi = 0;
        for (int c2 = 1; c2 < 256; ++c2) {
            float v = keybuf[t][c2];
            if (v < best) { best = v; bi = c2; }
        }
        long m = m0 + t;
        chunkVal[m * kMaxChunks + blockIdx.x] = best;
        chunkIdx[m * kMaxChunks + blockIdx.x] = (int)n0 + bi;
    }
}

// VQ stage 2: pick global min chunk, subtract code vector (f32), refresh bf16
__global__ void __launch_bounds__(256)
vq_update(float* __restrict__ resf, __bf16* __restrict__ resbf,
          const float* __restrict__ cb, const float* __restrict__ chunkVal,
          const int* __restrict__ chunkIdx, int nChunks,
          int* __restrict__ ids, int K) {
    __shared__ int sid;
    long m = blockIdx.x;
    if (threadIdx.x == 0) {
        float best = chunkVal[m * kMaxChunks];
        int bi = chunkIdx[m * kMaxChunks];
        for (int c = 1; c < nChunks; ++c) {
            float v = chunkVal[m * kMaxChunks + c];
            if (v < best) { best = v; bi = chunkIdx[m * kMaxChunks + c]; }
        }
        sid = bi;
        ids[m] = bi;
    }
    __syncthreads();
    int id = sid;
    const float* e = cb + (long)id * K;
    float* r  = resf + m * (long)K;
    __bf16* rb = resbf + m * (long)K;
    for (int i = threadIdx.x; i < K; i += 256) {
        float nv = r[i] - e[i];
        r[i] = nv;
        rb[i] = (__bf16)nv;
    }
}

// out[b,f,t] = ids[t][b*F+f] * (f < output_length[b])
__global__ void __launch_bounds__(256)
finalize(const int* __restrict__ ids, const int* __restrict__ outlen,
         int* __restrict__ out) {
    int idx = blockIdx.x * blockDim.x + threadIdx.x;
    if (idx >= kB * kF * 4) return;
    int t = idx & 3;
    int f = (idx >> 2) & (kF - 1);
    int b = idx >> 11;
    int valid = (f < outlen[b]) ? 1 : 0;
    out[idx] = valid ? ids[t * kM + b * kF + f] : 0;
}

// ---------------------------------------------------------------------------
// Host launcher
// ---------------------------------------------------------------------------
extern "C" void kernel_launch(void* const* d_in, const int* in_sizes, int n_in,
                              void* d_out, int out_size, void* d_ws, size_t ws_size,
                              hipStream_t stream) {
    (void)in_sizes; (void)n_in; (void)out_size; (void)ws_size;

    const float* x       = (const float*)d_in[0];   // [B,T,D] == [4096,5120]
    const int*   outlen  = (const int*)d_in[1];
    const float* w_gate  = (const float*)d_in[2];   // [I,D,P]
    const float* w_up    = (const float*)d_in[3];
    const float* w_down  = (const float*)d_in[4];   // [I,I]
    const float* ln_s    = (const float*)d_in[5];
    const float* ln_b    = (const float*)d_in[6];
    const float* cb[4]   = {(const float*)d_in[7], (const float*)d_in[8],
                            (const float*)d_in[9], (const float*)d_in[10]};
    const int cbsz[4]  = {8192, 8192, 4096, 4096};
    const int cboff[4] = {0, 8192, 16384, 20480};

    // workspace carve-up (256B aligned)
    char* ws = (char*)d_ws;
    size_t off = 0;
    auto carve = [&](size_t bytes) -> char* {
        char* p = ws + off;
        off = (off + bytes + 255) & ~(size_t)255;
        return p;
    };
    __bf16* xbf   = (__bf16*)carve((size_t)kM * kK * 2);
    __bf16* Bg    = (__bf16*)carve((size_t)kN * kK * 2);
    __bf16* Bu    = (__bf16*)carve((size_t)kN * kK * 2);
    __bf16* Bd    = (__bf16*)carve((size_t)kN * kK * 2);
    __bf16* cbbf  = (__bf16*)carve((size_t)kCBtot * kK * 2);
    float*  e2    = (float*) carve((size_t)kCBtot * 4);
    float*  gbuf  = (float*) carve((size_t)kM * kN * 4);  // g, later c
    float*  ubuf  = (float*) carve((size_t)kM * kN * 4);  // u, later residual f32
    __bf16* hbf   = (__bf16*)carve((size_t)kM * kK * 2);
    __bf16* resbf = (__bf16*)carve((size_t)kM * kK * 2);
    float*  chV   = (float*) carve((size_t)kM * kMaxChunks * 4);
    int*    chI   = (int*)   carve((size_t)kM * kMaxChunks * 4);
    int*    ids   = (int*)   carve((size_t)4 * kM * 4);

    const dim3 blk(256);

    // --- pack operands to bf16 WMMA layouts
    cvt_f32_bf16<<<4096, blk, 0, stream>>>(x, xbf, (long)kM * kK);
    pack_conv_w <<<4096, blk, 0, stream>>>(w_gate, Bg);
    pack_conv_w <<<4096, blk, 0, stream>>>(w_up,   Bu);
    cvt_f32_bf16<<<4096, blk, 0, stream>>>(w_down, Bd, (long)kN * kK);
    for (int t = 0; t < 4; ++t) {
        cvt_f32_bf16<<<4096, blk, 0, stream>>>(cb[t], cbbf + (long)cboff[t] * kK,
                                               (long)cbsz[t] * kK);
        row_norm2<<<cbsz[t], blk, 0, stream>>>(cb[t], e2 + cboff[t], kK);
    }

    // --- gated MLP: g = X Wg^T, u = X Wu^T, h = silu(g)*u, c = h Wd^T
    dim3 gemmGrid(kN / 256, kM / 32);
    gemm_bf16<<<gemmGrid, blk, 0, stream>>>(xbf, Bg, gbuf, kK, kN);
    gemm_bf16<<<gemmGrid, blk, 0, stream>>>(xbf, Bu, ubuf, kK, kN);
    silu_mul <<<4096, blk, 0, stream>>>(gbuf, ubuf, hbf, (long)kM * kN);
    gemm_bf16<<<gemmGrid, blk, 0, stream>>>(hbf, Bd, gbuf /*c*/, kK, kN);

    // --- residual add + LayerNorm -> residual (f32 in ubuf) + bf16 copy
    add_layernorm<<<kM, blk, 0, stream>>>(gbuf, x, ln_s, ln_b, ubuf, resbf, kK);

    // --- 4-stage residual VQ
    for (int t = 0; t < 4; ++t) {
        dim3 vqGrid(cbsz[t] / 256, kM / 32);
        vq_stage1<<<vqGrid, blk, 0, stream>>>(resbf, cbbf + (long)cboff[t] * kK,
                                              e2 + cboff[t], chV, chI, kK);
        vq_update<<<kM, blk, 0, stream>>>(ubuf, resbf, cb[t], chV, chI,
                                          cbsz[t] / 256, ids + t * kM, kK);
    }

    finalize<<<(kB * kF * 4 + 255) / 256, blk, 0, stream>>>(ids, outlen, (int*)d_out);
}
